// LNSLinear_90623809945932
// MI455X (gfx1250) — matrix operand order repaired and checked by reading, not compiled, not run on hardware
//
#include <hip/hip_runtime.h>

// y = x @ W^T + bias (the LNS log/exp machinery in the reference is an exact
// mathematical identity for the linear-domain matmul + bias).
//
// Two-pass split-precision bf16 GEMM for MI455X (gfx1250, wave32):
//
//  Pass 1 (convert_split_kernel): X,W (f32) -> Xhi,Xlo,Whi,Wlo (bf16) in d_ws.
//    The 3-term split  a*b ~= a_hi*b_hi + a_hi*b_lo + a_lo*b_hi  keeps ~16
//    mantissa bits/operand with f32 WMMA accumulation (~fp32-matmul accuracy)
//    at bf16-WMMA instruction rate (effective K=32/3 per inst vs K=4 for the
//    native f32 16x16x4 WMMA). Converting ONCE removes the v_bfe/v_add3
//    VALU storm from the GEMM hot loop (round-1 histogram: valu=538 vs wmma=9).
//
//  Pass 2 (gemm kernel): each wave owns a 32x32 output block (2x2 WMMA tiles).
//    Hot loop per 32-K step: 16x global_load_b128 (bf16, coalesced) feeding
//    12x v_wmma_f32_16x16x32_bf16, with only pointer math as VALU overhead.
//    Working set (~2 MB bf16) is fully L2-resident on the 192 MB L2.

typedef __attribute__((ext_vector_type(16))) __bf16 bf16x16;
typedef __attribute__((ext_vector_type(8)))  __bf16 bf16x8;
typedef __attribute__((ext_vector_type(4)))  __bf16 bf16x4;
typedef __attribute__((ext_vector_type(8)))  float  f32x8;

static __device__ __forceinline__ __bf16 bf_from_bits(unsigned short u) {
  return __builtin_bit_cast(__bf16, u);
}

// Split f32 into hi (truncated bf16, so f - hi is exact) + lo (RNE bf16).
static __device__ __forceinline__ void split_bf16(float f, __bf16& hi, __bf16& lo) {
  unsigned u  = __builtin_bit_cast(unsigned, f);
  unsigned short hb = (unsigned short)(u >> 16);
  float hf = __builtin_bit_cast(float, u & 0xFFFF0000u);
  float lf = f - hf;
  unsigned lu = __builtin_bit_cast(unsigned, lf);
  unsigned lr = lu + 0x7FFFu + ((lu >> 16) & 1u);   // round-to-nearest-even
  hi = bf_from_bits(hb);
  lo = bf_from_bits((unsigned short)(lr >> 16));
}

static __device__ __forceinline__ bf16x16 cat8(bf16x8 a, bf16x8 b) {
  return __builtin_shufflevector(a, b, 0, 1, 2, 3, 4, 5, 6, 7,
                                       8, 9, 10, 11, 12, 13, 14, 15);
}

// ---------------- Pass 1: one-time f32 -> (hi, lo) bf16 split ---------------
__global__ __launch_bounds__(256)
void convert_split_kernel(const float* __restrict__ src,
                          __bf16* __restrict__ hi,
                          __bf16* __restrict__ lo, int n)
{
  const int i = (blockIdx.x * 256 + threadIdx.x) * 4;
  if (i >= n) return;
  const float4 v = *(const float4*)(src + i);
  bf16x4 h, l;
  __bf16 th, tl;
  split_bf16(v.x, th, tl); h[0] = th; l[0] = tl;
  split_bf16(v.y, th, tl); h[1] = th; l[1] = tl;
  split_bf16(v.z, th, tl); h[2] = th; l[2] = tl;
  split_bf16(v.w, th, tl); h[3] = th; l[3] = tl;
  *(bf16x4*)(hi + i) = h;
  *(bf16x4*)(lo + i) = l;
}

// ---------------- Pass 2: 2x2-tile WMMA GEMM with bias epilogue -------------
__global__ __launch_bounds__(128)
void lns_linear_wmma_gemm(const __bf16* __restrict__ Xhi,
                          const __bf16* __restrict__ Xlo,
                          const __bf16* __restrict__ Whi,
                          const __bf16* __restrict__ Wlo,
                          const float*  __restrict__ bias,
                          float* __restrict__ Y,
                          int K, int N, int tilesN2, int totalTiles2)
{
  const int lane = threadIdx.x & 31;
  const int wave = threadIdx.x >> 5;
  const int tile = blockIdx.x * 4 + wave;
  if (tile >= totalTiles2) return;             // wave-uniform exit

  const int tm = (tile / tilesN2) << 5;        // 32-row output base
  const int tn = (tile % tilesN2) << 5;        // 32-col output base

  const int half = lane >> 4;                  // 0: lanes 0-15, 1: lanes 16-31
  const int l15  = lane & 15;

  // A fragment (16x32 bf16): lane row m=l15; elems 0..7 -> K=kb..kb+7,
  // elems 8..15 -> K=16+kb..16+kb+7, kb = 8*half.
  const int aoff0 = (tm +      l15) * K + half * 8;
  const int aoff1 = (tm + 16 + l15) * K + half * 8;
  // B fragment (32x16): lane column n=l15; same K order as A (kb, 16+kb).
  const int boff0 = (tn +      l15) * K + half * 8;
  const int boff1 = (tn + 16 + l15) * K + half * 8;

  f32x8 acc00 = {}, acc01 = {}, acc10 = {}, acc11 = {};

  #pragma unroll 2
  for (int k0 = 0; k0 < K; k0 += 32) {
    // 16-byte (8 x bf16) coalesced loads; all offsets 16B-aligned.
    const bf16x16 a0h = cat8(*(const bf16x8*)(Xhi + aoff0 + k0),
                             *(const bf16x8*)(Xhi + aoff0 + k0 + 16));
    const bf16x16 a0l = cat8(*(const bf16x8*)(Xlo + aoff0 + k0),
                             *(const bf16x8*)(Xlo + aoff0 + k0 + 16));
    const bf16x16 a1h = cat8(*(const bf16x8*)(Xhi + aoff1 + k0),
                             *(const bf16x8*)(Xhi + aoff1 + k0 + 16));
    const bf16x16 a1l = cat8(*(const bf16x8*)(Xlo + aoff1 + k0),
                             *(const bf16x8*)(Xlo + aoff1 + k0 + 16));
    const bf16x16 b0h = cat8(*(const bf16x8*)(Whi + boff0 + k0),
                             *(const bf16x8*)(Whi + boff0 + k0 + 16));
    const bf16x16 b0l = cat8(*(const bf16x8*)(Wlo + boff0 + k0),
                             *(const bf16x8*)(Wlo + boff0 + k0 + 16));
    const bf16x16 b1h = cat8(*(const bf16x8*)(Whi + boff1 + k0),
                             *(const bf16x8*)(Whi + boff1 + k0 + 16));
    const bf16x16 b1l = cat8(*(const bf16x8*)(Wlo + boff1 + k0),
                             *(const bf16x8*)(Wlo + boff1 + k0 + 16));

    // 12 WMMAs per K-step: hi*hi + hi*lo + lo*hi for each of the 2x2 tiles.
    acc00 = __builtin_amdgcn_wmma_f32_16x16x32_bf16(false, a0h, false, b0h, (short)0, acc00, false, false);
    acc00 = __builtin_amdgcn_wmma_f32_16x16x32_bf16(false, a0h, false, b0l, (short)0, acc00, false, false);
    acc00 = __builtin_amdgcn_wmma_f32_16x16x32_bf16(false, a0l, false, b0h, (short)0, acc00, false, false);

    acc01 = __builtin_amdgcn_wmma_f32_16x16x32_bf16(false, a0h, false, b1h, (short)0, acc01, false, false);
    acc01 = __builtin_amdgcn_wmma_f32_16x16x32_bf16(false, a0h, false, b1l, (short)0, acc01, false, false);
    acc01 = __builtin_amdgcn_wmma_f32_16x16x32_bf16(false, a0l, false, b1h, (short)0, acc01, false, false);

    acc10 = __builtin_amdgcn_wmma_f32_16x16x32_bf16(false, a1h, false, b0h, (short)0, acc10, false, false);
    acc10 = __builtin_amdgcn_wmma_f32_16x16x32_bf16(false, a1h, false, b0l, (short)0, acc10, false, false);
    acc10 = __builtin_amdgcn_wmma_f32_16x16x32_bf16(false, a1l, false, b0h, (short)0, acc10, false, false);

    acc11 = __builtin_amdgcn_wmma_f32_16x16x32_bf16(false, a1h, false, b1h, (short)0, acc11, false, false);
    acc11 = __builtin_amdgcn_wmma_f32_16x16x32_bf16(false, a1h, false, b1l, (short)0, acc11, false, false);
    acc11 = __builtin_amdgcn_wmma_f32_16x16x32_bf16(false, a1l, false, b1h, (short)0, acc11, false, false);
  }

  // C/D layout: VGPR r, lane L -> row = 8*(L>=16)+r, col = L&15.
  const int col0 = tn + l15;
  const int col1 = tn + 16 + l15;
  const float bv0 = bias[col0];
  const float bv1 = bias[col1];
  #pragma unroll
  for (int r = 0; r < 8; ++r) {
    const int row0 = tm + half * 8 + r;
    const int row1 = row0 + 16;
    Y[(size_t)row0 * N + col0] = acc00[r] + bv0;
    Y[(size_t)row0 * N + col1] = acc01[r] + bv1;
    Y[(size_t)row1 * N + col0] = acc10[r] + bv0;
    Y[(size_t)row1 * N + col1] = acc11[r] + bv1;
  }
}

extern "C" void kernel_launch(void* const* d_in, const int* in_sizes, int n_in,
                              void* d_out, int out_size, void* d_ws, size_t ws_size,
                              hipStream_t stream) {
  const float* X    = (const float*)d_in[0];   // [B,K]
  const float* W    = (const float*)d_in[1];   // [N,K]
  const float* bias = (const float*)d_in[2];   // [N]
  float* Y = (float*)d_out;

  const int N = in_sizes[2];
  const int K = in_sizes[1] / N;
  const int B = in_sizes[0] / K;
  const int nX = B * K;
  const int nW = N * K;

  // Workspace layout (bf16): Xhi | Xlo | Whi | Wlo   (4*(nX+nW) bytes total).
  __bf16* Xhi = (__bf16*)d_ws;
  __bf16* Xlo = Xhi + nX;
  __bf16* Whi = Xlo + nX;
  __bf16* Wlo = Whi + nW;

  // Pass 1: one-time f32 -> (hi,lo) bf16 split.
  convert_split_kernel<<<(nX / 4 + 255) / 256, 256, 0, stream>>>(X, Xhi, Xlo, nX);
  convert_split_kernel<<<(nW / 4 + 255) / 256, 256, 0, stream>>>(W, Whi, Wlo, nW);

  // Pass 2: 2x2-tile WMMA GEMM (stream ordering makes pass 1 visible).
  const int tilesM2 = B >> 5;
  const int tilesN2 = N >> 5;
  const int totalTiles2 = tilesM2 * tilesN2;       // 256 for 512x512
  const int blocks = (totalTiles2 + 3) / 4;        // 4 waves (128 thr) / block

  lns_linear_wmma_gemm<<<blocks, 128, 0, stream>>>(Xhi, Xlo, Whi, Wlo, bias, Y,
                                                   K, N, tilesN2, totalTiles2);
}
